// MultiViewDualSoftmaxLoss_31164282699995
// MI455X (gfx1250) — matrix operand order repaired and checked by reading, not compiled, not run on hardware
//
#include <hip/hip_runtime.h>
#include <hip/hip_bf16.h>

typedef __attribute__((ext_vector_type(16))) __bf16 v16bf;
typedef __attribute__((ext_vector_type(8)))  __bf16 v8bf;
typedef __attribute__((ext_vector_type(8)))  float  v8f;

#define NTOK   4096
#define CDIM   128
#define NVIEW  4
#define NPAIRS 6

// ---- order-preserving float <-> uint encoding for atomicMax on floats ----
__device__ __forceinline__ unsigned encOrd(float f) {
    unsigned u = __float_as_uint(f);
    return (u & 0x80000000u) ? ~u : (u | 0x80000000u);
}
__device__ __forceinline__ float decOrd(unsigned u) {
    u = (u & 0x80000000u) ? (u & 0x7fffffffu) : ~u;
    return __uint_as_float(u);
}
__device__ __forceinline__ float get_temp(const float* lt) {
    return fminf(fmaxf(expf(lt[0]), 0.01f), 10.0f);
}

// CDNA5 async global->LDS copy, 16 bytes per lane. vdst = LDS byte offset
// (low 32 bits of the flat shared-aperture address), vaddr = 64-bit global.
// Tracked by ASYNCcnt.
__device__ __forceinline__ void async_ld16(void* lds, const void* g) {
    unsigned loff = (unsigned)(uintptr_t)lds;  // shared aperture: addr[31:0] == LDS offset
    asm volatile("global_load_async_to_lds_b128 %0, %1, off"
                 :: "v"(loff), "v"(g)
                 : "memory");
}
__device__ __forceinline__ void wait_async_le8() {
    asm volatile("s_wait_asynccnt 8" ::: "memory");
}
__device__ __forceinline__ void wait_async_0() {
    asm volatile("s_wait_asynccnt 0" ::: "memory");
}

// ---------------------------------------------------------------------------
// Kernel 0: zero all stats (encoded max identity == 0 bits) and the loss slot.
// stats layout (contiguous): rowsum | colsum | diag | rowmaxU | colmaxU,
// each NPAIRS*NTOK 32-bit words.
// ---------------------------------------------------------------------------
__global__ void init_kernel(unsigned* __restrict__ stats, float* __restrict__ out) {
    int t = blockIdx.x * blockDim.x + threadIdx.x;
    if (t < 5 * NPAIRS * NTOK) stats[t] = 0u;
    if (t == 0) out[0] = 0.0f;
}

// ---------------------------------------------------------------------------
// Kernel 1: L2-normalize each (n,v) row of desc, store bf16 in (V,N,C) layout.
// One wave32 per row; lane holds 4 channels.
// ---------------------------------------------------------------------------
__global__ __launch_bounds__(256) void normalize_kernel(const float* __restrict__ desc,
                                                        __bf16* __restrict__ dn) {
    int row  = (blockIdx.x * blockDim.x + threadIdx.x) >> 5;  // 0..16383 == n*V+v
    int lane = threadIdx.x & 31;
    const float4 x = ((const float4*)(desc + (size_t)row * CDIM))[lane];
    float ss = x.x * x.x + x.y * x.y + x.z * x.z + x.w * x.w;
    #pragma unroll
    for (int m = 1; m < 32; m <<= 1) ss += __shfl_xor(ss, m, 32);
    float inv = 1.0f / fmaxf(sqrtf(ss), 1e-12f);
    int n = row >> 2, v = row & 3;
    __bf16 o[4];
    o[0] = (__bf16)(x.x * inv);
    o[1] = (__bf16)(x.y * inv);
    o[2] = (__bf16)(x.z * inv);
    o[3] = (__bf16)(x.w * inv);
    __bf16* dst = dn + ((size_t)v * NTOK + n) * CDIM + lane * 4;
    *(uint2*)dst = *(uint2*)o;  // 8-byte store
}

// ---------------------------------------------------------------------------
// Kernel 2: fused bf16 WMMA GEMM + single-pass softmax statistics.
// grid = (32*32 tiles, NPAIRS), block = 256 (8 waves).
// Workgroup tile 128x128; wave tile 32x64 = 2x4 accumulators of 16x16 f32.
// Both K=64 stages prefetched up-front with async global->LDS into double
// buffers (~74 KB LDS, fine on CDNA5's 320 KB WGP LDS); stage-1 loads overlap
// stage-0 WMMA compute.
// ---------------------------------------------------------------------------
__global__ __launch_bounds__(256) void pair_kernel(
    const __bf16* __restrict__ dn, const float* __restrict__ logtemp,
    float* __restrict__ rowsum, unsigned* __restrict__ rowmaxU,
    float* __restrict__ colsum, unsigned* __restrict__ colmaxU,
    float* __restrict__ diagv) {
    // padded LDS tiles: 72*2 = 144B row stride (not a multiple of 256B -> fragment
    // gathers hit rotating bank groups instead of 16-way conflicts)
    __shared__ __align__(16) __bf16 smA[2][128][72];
    __shared__ __align__(16) __bf16 smB[2][128][72];

    const int piv[NPAIRS] = {0, 0, 0, 1, 1, 2};
    const int pjv[NPAIRS] = {1, 2, 3, 2, 3, 3};
    const int pair    = blockIdx.y;
    const int rowBase = (blockIdx.x >> 5) * 128;
    const int colBase = (blockIdx.x & 31) * 128;
    const size_t vs   = (size_t)NTOK * CDIM;
    const __bf16* dA  = dn + piv[pair] * vs;  // rows (i view)
    const __bf16* dB  = dn + pjv[pair] * vs;  // cols (j view)

    const int t    = threadIdx.x;
    const int lane = t & 31;
    const int wave = t >> 5;
    const int wr   = wave >> 1;   // 0..3 -> 32-row slice
    const int wc   = wave & 1;    // 0..1 -> 64-col slice
    const int half = lane >> 4;   // 0/1
    const int l16  = lane & 15;

    const float temp = get_temp(logtemp);

    v8f acc[2][4];
    #pragma unroll
    for (int mt = 0; mt < 2; ++mt)
        #pragma unroll
        for (int nt = 0; nt < 4; ++nt) acc[mt][nt] = (v8f){};

    // ---- async staging: thread t covers 64B (4 x b128) of one row-half per
    // matrix per stage. Stage s handles K block [64s, 64s+64).
    const int ldRow = t >> 1;           // 0..127
    const int ldSeg = (t & 1) * 32;     // 0 or 32 (bf16 elems; 64B)
    {
        const __bf16* gA = dA + (size_t)(rowBase + ldRow) * CDIM + ldSeg;
        const __bf16* gB = dB + (size_t)(colBase + ldRow) * CDIM + ldSeg;
        #pragma unroll
        for (int s = 0; s < 2; ++s) {    // 8 async instructions per stage per wave
            #pragma unroll
            for (int q = 0; q < 4; ++q) {
                async_ld16(&smA[s][ldRow][ldSeg + 8 * q], gA + s * 64 + 8 * q);
                async_ld16(&smB[s][ldRow][ldSeg + 8 * q], gB + s * 64 + 8 * q);
            }
        }
    }

    auto compute_stage = [&](int s) {
        #pragma unroll
        for (int ks = 0; ks < 64; ks += 32) {
            // A fragments: lane l<16 holds M=l, K {ks..ks+7, ks+16..ks+23};
            //              lane l>=16 holds M=l-16, K {ks+8..ks+15, ks+24..ks+31}
            v16bf afrag[2];
            #pragma unroll
            for (int mt = 0; mt < 2; ++mt) {
                int mrow = wr * 32 + mt * 16 + l16;
                v8bf a0 = *(const v8bf*)&smA[s][mrow][ks + half * 8];
                v8bf a1 = *(const v8bf*)&smA[s][mrow][ks + 16 + half * 8];
                #pragma unroll
                for (int e = 0; e < 8; ++e) { afrag[mt][e] = a0[e]; afrag[mt][e + 8] = a1[e]; }
            }
            // B fragments (B = d_j^T): same K layout as A, l16 selects N.
            v16bf bfrag[4];
            #pragma unroll
            for (int nt = 0; nt < 4; ++nt) {
                int nrow = wc * 64 + nt * 16 + l16;
                v8bf b0 = *(const v8bf*)&smB[s][nrow][ks + half * 8];
                v8bf b1 = *(const v8bf*)&smB[s][nrow][ks + 16 + half * 8];
                #pragma unroll
                for (int e = 0; e < 8; ++e) { bfrag[nt][e] = b0[e]; bfrag[nt][e + 8] = b1[e]; }
            }
            #pragma unroll
            for (int mt = 0; mt < 2; ++mt)
                #pragma unroll
                for (int nt = 0; nt < 4; ++nt)
                    acc[mt][nt] = __builtin_amdgcn_wmma_f32_16x16x32_bf16(
                        false, afrag[mt], false, bfrag[nt], (short)0, acc[mt][nt],
                        false, false);
        }
    };

    wait_async_le8();   // stage-0 (first 8 per-wave async ops) complete
    __syncthreads();    // ... for all waves
    compute_stage(0);   // overlaps with stage-1 loads still in flight
    wait_async_0();     // stage-1 complete
    __syncthreads();
    compute_stage(1);

    // ---- epilogue: s = dot*temp; e = exp(s - temp) (fixed stable shift since |s|<=temp)
    // C/D layout: VGPR r -> M = r (lanes 0-15) or 8+r (lanes 16-31); N = l16.
    const int po = pair * NTOK;
    float colS[4] = {0.f, 0.f, 0.f, 0.f};
    float colM[4] = {-3e38f, -3e38f, -3e38f, -3e38f};
    #pragma unroll
    for (int mt = 0; mt < 2; ++mt) {
        #pragma unroll
        for (int r = 0; r < 8; ++r) {
            const int gm = rowBase + wr * 32 + mt * 16 + half * 8 + r;
            float rS = 0.f, rM = -3e38f;
            #pragma unroll
            for (int nt = 0; nt < 4; ++nt) {
                float sv = acc[mt][nt][r] * temp;
                float e = __expf(sv - temp);
                rS += e;
                rM = fmaxf(rM, sv);
                colS[nt] += e;
                colM[nt] = fmaxf(colM[nt], sv);
                int gn = colBase + wc * 64 + nt * 16 + l16;
                if (gm == gn) diagv[po + gm] = sv;  // unique writer
            }
            // reduce across the 16 lanes sharing this row (stay inside each half)
            #pragma unroll
            for (int m = 1; m < 16; m <<= 1) {
                rS += __shfl_xor(rS, m, 32);
                rM = fmaxf(rM, __shfl_xor(rM, m, 32));
            }
            if (l16 == 0) {
                atomicAdd(&rowsum[po + gm], rS);
                atomicMax(&rowmaxU[po + gm], encOrd(rM));
            }
        }
    }
    #pragma unroll
    for (int nt = 0; nt < 4; ++nt) {
        float cS = colS[nt] + __shfl_xor(colS[nt], 16, 32);
        float cM = fmaxf(colM[nt], __shfl_xor(colM[nt], 16, 32));
        if (half == 0) {
            int gn = colBase + wc * 64 + nt * 16 + l16;
            atomicAdd(&colsum[po + gn], cS);
            atomicMax(&colmaxU[po + gn], encOrd(cM));
        }
    }
}

// ---------------------------------------------------------------------------
// Kernel 3: assemble loss + conf from statistics.
// out[0] = loss; out[1 + n*4 + v] = conf[n][v]
// ---------------------------------------------------------------------------
__global__ __launch_bounds__(256) void finalize_kernel(
    const float* __restrict__ rowsum, const unsigned* __restrict__ rowmaxU,
    const float* __restrict__ colsum, const unsigned* __restrict__ colmaxU,
    const float* __restrict__ diagv, const float* __restrict__ logtemp,
    float* __restrict__ out) {
    int n = blockIdx.x * blockDim.x + threadIdx.x;
    if (n >= NTOK) return;
    const int piv[NPAIRS] = {0, 0, 0, 1, 1, 2};
    const int pjv[NPAIRS] = {1, 2, 3, 2, 3, 3};
    const float temp = get_temp(logtemp);
    float conf[NVIEW] = {0.f, 0.f, 0.f, 0.f};
    float lossAcc = 0.f;
    #pragma unroll
    for (int p = 0; p < NPAIRS; ++p) {
        int o = p * NTOK + n;
        float lse_r = temp + logf(rowsum[o]);   // log sum exp over row
        float lse_c = temp + logf(colsum[o]);   // log sum exp over column
        float dg    = diagv[o];
        conf[piv[p]] += expf(decOrd(rowmaxU[o]) - lse_r);  // max row prob
        conf[pjv[p]] += expf(decOrd(colmaxU[o]) - lse_c);
        lossAcc += (lse_r - dg) + (lse_c - dg);            // -log p on diagonal
    }
    #pragma unroll
    for (int v = 0; v < NVIEW; ++v)
        out[1 + n * NVIEW + v] = conf[v] * (1.0f / (NVIEW - 1));
    atomicAdd(out, lossAcc * (1.0f / (NPAIRS * (float)NTOK)));
}

// ---------------------------------------------------------------------------
extern "C" void kernel_launch(void* const* d_in, const int* in_sizes, int n_in,
                              void* d_out, int out_size, void* d_ws, size_t ws_size,
                              hipStream_t stream) {
    const float* desc    = (const float*)d_in[0];
    // d_in[1] (var) unused by the reference math
    const float* logtemp = (const float*)d_in[2];
    float* out = (float*)d_out;

    char* ws = (char*)d_ws;
    __bf16* dn = (__bf16*)ws;                                   // 4 MB normalized bf16 (V,N,C)
    size_t dnBytes = (size_t)NVIEW * NTOK * CDIM * sizeof(__bf16);
    float*    rowsum  = (float*)(ws + dnBytes);                 // 5 contiguous stat arrays
    float*    colsum  = rowsum + NPAIRS * NTOK;
    float*    diagv   = colsum + NPAIRS * NTOK;
    unsigned* rowmaxU = (unsigned*)(diagv + NPAIRS * NTOK);
    unsigned* colmaxU = rowmaxU + NPAIRS * NTOK;

    init_kernel<<<(5 * NPAIRS * NTOK + 255) / 256, 256, 0, stream>>>((unsigned*)rowsum, out);
    normalize_kernel<<<(NVIEW * NTOK * 32) / 256, 256, 0, stream>>>(desc, dn);
    dim3 grid(32 * 32, NPAIRS);
    pair_kernel<<<grid, 256, 0, stream>>>(dn, logtemp, rowsum, rowmaxU, colsum, colmaxU, diagv);
    finalize_kernel<<<NTOK / 256, 256, 0, stream>>>(rowsum, rowmaxU, colsum, colmaxU, diagv,
                                                    logtemp, out);
}
